// GCMC_84026740179237
// MI455X (gfx1250) — matrix operand order repaired and checked by reading, not compiled, not run on hardware
//
#include <hip/hip_runtime.h>
#include <hip/hip_bf16.h>

// ---------------------------------------------------------------------------
// GCMC forward on gfx1250: bf16 WMMA GEMMs + wave-per-edge f32 atomic scatter.
// ---------------------------------------------------------------------------

#define CLASSES   5
#define D_FEAT    256
#define D_CONV    128
#define D_RAW     256
#define D_EMB     256
#define H_DIM     (CLASSES * D_CONV)   // 640
#define BN_EPS    1e-3f

typedef __attribute__((ext_vector_type(16))) __bf16          v16bf;
typedef __attribute__((ext_vector_type(16))) unsigned short  v16u;
typedef __attribute__((ext_vector_type(8)))  float           v8f;

enum { GF_RELU = 1, GF_ACC = 2 };

__device__ __forceinline__ unsigned short f2bf(float f) {
    unsigned u = __builtin_bit_cast(unsigned, f);
    u += 0x7FFFu + ((u >> 16) & 1u);         // round to nearest even
    return (unsigned short)(u >> 16);
}
__device__ __forceinline__ unsigned pack2bf(float lo, float hi) {
    return (unsigned)f2bf(lo) | ((unsigned)f2bf(hi) << 16);
}

// ---------------------------------------------------------------------------
// Generic tiled GEMM:  C[M,N] (+)= act( scale .* (A[M,K] @ B[K,N]) + shift )
//   - A rows optionally gathered through rowidx (embedding lookup fusion)
//   - f32 global -> packed-bf16 LDS staging -> v_wmma_f32_16x16x32_bf16
//   - block tile 128x128, 8 waves; each wave owns a 32x64 strip:
//       2 A-frags x 4 B-frags = 8 WMMAs per K-step (register reuse 2x/4x)
// Requires: M % 128 == 0, N % 128 == 0, K % 32 == 0.
// ---------------------------------------------------------------------------
#define BM 128
#define BN 128
#define BK 32
#define SAS 48   // padded LDS row stride (bf16 elems); 96B keeps 16B alignment
#define SBS 48

__global__ __launch_bounds__(256)
void gemm_bf16_wmma(const float* __restrict__ A, int lda,
                    const int* __restrict__ rowidx,
                    const float* __restrict__ B, int ldb,
                    float* __restrict__ C, int ldc,
                    int K,
                    const float* __restrict__ scale,
                    const float* __restrict__ shift,
                    int flags)
{
    __shared__ __align__(64) unsigned short sA[BM * SAS];
    __shared__ __align__(64) unsigned short sB[BN * SBS];  // transposed: sB[n][k]

    const int tid   = threadIdx.x;
    const int lane  = tid & 31;
    const int wave  = tid >> 5;
    const int m_off = (wave & 3) * 32;    // 4 waves along M, 32 rows each
    const int n_off = (wave >> 2) * 64;   // 2 waves along N, 4 subtiles each

    const long long blockM = (long long)blockIdx.y * BM;
    const long long blockN = (long long)blockIdx.x * BN;

    v8f acc[2][4];
    const v8f vzero = {0.f, 0.f, 0.f, 0.f, 0.f, 0.f, 0.f, 0.f};
#pragma unroll
    for (int mt = 0; mt < 2; ++mt)
#pragma unroll
        for (int nt = 0; nt < 4; ++nt) acc[mt][nt] = vzero;

    // A loader: 128 rows x 32 K, 16 K-elems per thread (2 threads per row)
    const int ar = tid >> 1;
    const int ak = (tid & 1) * 16;
    const long long a_row = blockM + ar;
    const long long a_src = rowidx ? (long long)rowidx[a_row] : a_row;
    const float* Abase = A + a_src * (long long)lda + ak;

    // B loader: 32 K-rows x 128 N; each thread covers 8 N x 2 K (packed b32)
    const int bkp = tid >> 4;           // 0..15 : K-pair index
    const int bn  = (tid & 15) * 8;     // 0..120
    const float* Bbase = B + (long long)(2 * bkp) * ldb + blockN + bn;

    const int fr = lane & 15;   // fragment row/col within 16
    const int fh = lane >> 4;   // fragment K-half selector

    for (int k0 = 0; k0 < K; k0 += BK) {
        {   // stage A tile: 16 floats -> 2 x ds_store_b128 of packed bf16
            const float* p = Abase + k0;
            uint4 w0, w1;
            w0.x = pack2bf(p[0],  p[1]);  w0.y = pack2bf(p[2],  p[3]);
            w0.z = pack2bf(p[4],  p[5]);  w0.w = pack2bf(p[6],  p[7]);
            w1.x = pack2bf(p[8],  p[9]);  w1.y = pack2bf(p[10], p[11]);
            w1.z = pack2bf(p[12], p[13]); w1.w = pack2bf(p[14], p[15]);
            uint4* d = (uint4*)&sA[ar * SAS + ak];
            d[0] = w0; d[1] = w1;
        }
        {   // stage B tile transposed: pack (k, k+1) pairs -> b32 stores
            const float* p0 = Bbase + (long long)k0 * ldb;
            const float* p1 = p0 + ldb;
#pragma unroll
            for (int j = 0; j < 8; ++j) {
                *(unsigned*)&sB[(bn + j) * SBS + 2 * bkp] = pack2bf(p0[j], p1[j]);
            }
        }
        __syncthreads();

        // A fragments (ISA 7.12.2 16-bit A layout: lanes 0-15 hold K 0-7 &
        // 16-23; lanes 16-31 hold K 8-15 & 24-31)
        v16bf afrag[2];
#pragma unroll
        for (int mt = 0; mt < 2; ++mt) {
            v16u a_us;
            const unsigned short* base = &sA[(m_off + mt * 16 + fr) * SAS];
#pragma unroll
            for (int e = 0; e < 8; ++e) a_us[e]     = base[fh * 8 + e];
#pragma unroll
            for (int e = 0; e < 8; ++e) a_us[8 + e] = base[16 + fh * 8 + e];
            afrag[mt] = __builtin_bit_cast(v16bf, a_us);
        }

#pragma unroll
        for (int nt = 0; nt < 4; ++nt) {
            // B fragment: col N = lane%16, same K layout as A (K 8h..8h+7 & 16+8h..16+8h+7)
            v16u b_us;
            const unsigned short* base = &sB[(n_off + nt * 16 + fr) * SBS];
#pragma unroll
            for (int e = 0; e < 16; ++e) b_us[e] = base[(e & 8) * 2 + fh * 8 + (e & 7)];
            const v16bf bfrag = __builtin_bit_cast(v16bf, b_us);

#pragma unroll
            for (int mt = 0; mt < 2; ++mt) {
                acc[mt][nt] = __builtin_amdgcn_wmma_f32_16x16x32_bf16(
                    false, afrag[mt], false, bfrag, (short)0, acc[mt][nt],
                    false, false);
            }
        }
        __syncthreads();
    }

    // Epilogue. C/D layout: lane holds N = lane%16, M = v + 8*(lane/16).
#pragma unroll
    for (int nt = 0; nt < 4; ++nt) {
        const long long n_g = blockN + n_off + nt * 16 + fr;
        const float sc = scale ? scale[n_g] : 1.f;  // indexed by GEMM column
        const float sh = shift ? shift[n_g] : 0.f;
#pragma unroll
        for (int mt = 0; mt < 2; ++mt) {
#pragma unroll
            for (int v = 0; v < 8; ++v) {
                const long long m_g = blockM + m_off + mt * 16 + fh * 8 + v;
                float* cp = C + m_g * (long long)ldc + n_g;
                float x = acc[mt][nt][v];
                if (flags & GF_ACC) x += *cp;
                x = x * sc + sh;
                if (flags & GF_RELU) x = fmaxf(x, 0.f);
                *cp = x;
            }
        }
    }
}

// ---------------------------------------------------------------------------
// Weighted edge aggregation: out[c*B + dst] += feat[src] * w
// Two edges per wave; prefetch the 2nd edge's 1KB feature row into L2/L0
// (global_prefetch_b8) while scatter-adding the 1st.
// ---------------------------------------------------------------------------
__global__ __launch_bounds__(256)
void edge_agg(const float* __restrict__ feat,
              const int* __restrict__ src, const int* __restrict__ dst,
              const float* __restrict__ w,
              float* __restrict__ out, int E_per_class, int Bsz,
              long long total_edges)
{
    const long long pair = (long long)blockIdx.x * (blockDim.x >> 5) + (threadIdx.x >> 5);
    const long long e0 = pair * 2;
    if (e0 >= total_edges) return;
    const int lane = threadIdx.x & 31;

    const long long e1 = e0 + 1;
    const bool has1 = (e1 < total_edges);
    const int s1 = has1 ? src[e1] : 0;
    if (has1)   // warm L2/WGP$ for the next edge's row while we work on e0
        __builtin_prefetch(feat + (long long)s1 * D_FEAT + lane * 8, 0, 1);

#pragma unroll 1
    for (int t = 0; t < 2; ++t) {
        const long long e = e0 + t;
        if (e >= total_edges) break;
        const int s  = t ? s1 : src[e];
        const int c  = (int)(e / E_per_class);
        const int d  = dst[e];
        const float ww = w[e];

        const float4* fp = (const float4*)(feat + (long long)s * D_FEAT + lane * 8);
        float* o = out + ((long long)c * Bsz + d) * (long long)D_FEAT + lane * 8;
        const float4 f0 = fp[0];
        const float4 f1 = fp[1];
        atomicAdd(&o[0], f0.x * ww); atomicAdd(&o[1], f0.y * ww);
        atomicAdd(&o[2], f0.z * ww); atomicAdd(&o[3], f0.w * ww);
        atomicAdd(&o[4], f1.x * ww); atomicAdd(&o[5], f1.y * ww);
        atomicAdd(&o[6], f1.z * ww); atomicAdd(&o[7], f1.w * ww);
    }
}

// ---------------------------------------------------------------------------
// BN(+bias) fold: scale = g*rsqrt(v+eps), shift = scale*(bias-mean)+beta
// ---------------------------------------------------------------------------
__global__ void bn_prep(const float* __restrict__ bn, const float* __restrict__ bias,
                        int dim, float* __restrict__ scale, float* __restrict__ shift)
{
    const int i = blockIdx.x * blockDim.x + threadIdx.x;
    if (i >= dim) return;
    const float g = bn[i], b = bn[dim + i], m = bn[2 * dim + i], v = bn[3 * dim + i];
    const float sc = g * rsqrtf(v + BN_EPS);
    const float bi = bias ? bias[i] : 0.f;
    scale[i] = sc;
    shift[i] = sc * (bi - m) + b;
}

__global__ void zero_fill(float* __restrict__ p, long long n)
{
    long long i = (long long)blockIdx.x * blockDim.x + threadIdx.x;
    const long long stride = (long long)gridDim.x * blockDim.x;
    for (; i < n; i += stride) p[i] = 0.f;
}

// kern[r] = Wcomb[r,0]*Wdec[0] + Wcomb[r,1]*Wdec[1]
__global__ void dec_combine(const float* __restrict__ Wdec, const float* __restrict__ Wcomb,
                            float* __restrict__ kern)
{
    const int i = blockIdx.x * blockDim.x + threadIdx.x;
    if (i >= CLASSES * D_EMB * D_EMB) return;
    const int r  = i >> 16;           // D_EMB*D_EMB == 65536
    const int ij = i & 65535;
    kern[i] = Wcomb[r * 2] * Wdec[ij] + Wcomb[r * 2 + 1] * Wdec[65536 + ij];
}

// logits[b, r] = dot(tmp[r][b], item_emb[b])   (one wave per (r,b))
__global__ __launch_bounds__(256)
void bilinear_dot(const float* __restrict__ tmp, const float* __restrict__ iemb,
                  float* __restrict__ out, int Bsz)
{
    const long long gw = (long long)blockIdx.x * (blockDim.x >> 5) + (threadIdx.x >> 5);
    if (gw >= (long long)CLASSES * Bsz) return;
    const int lane = threadIdx.x & 31;
    const int r = (int)(gw / Bsz);
    const int b = (int)(gw % Bsz);
    const float* t = tmp  + ((long long)r * Bsz + b) * D_EMB;
    const float* v = iemb + (long long)b * D_EMB;
    float s = 0.f;
#pragma unroll
    for (int j = 0; j < D_EMB / 32; ++j) s += t[lane + j * 32] * v[lane + j * 32];
#pragma unroll
    for (int off = 16; off > 0; off >>= 1) s += __shfl_xor(s, off, 32);
    if (lane == 0) out[(long long)b * CLASSES + r] = s;
}

// ---------------------------------------------------------------------------
extern "C" void kernel_launch(void* const* d_in, const int* in_sizes, int n_in,
                              void* d_out, int out_size, void* d_ws, size_t ws_size,
                              hipStream_t stream)
{
    const float* user_feat = (const float*)d_in[0];
    const float* item_feat = (const float*)d_in[1];
    const int*   user_idx  = (const int*)d_in[2];
    const int*   item_idx  = (const int*)d_in[3];
    const int*   u_src = (const int*)d_in[4];
    const int*   u_dst = (const int*)d_in[5];
    const float* u_w   = (const float*)d_in[6];
    const int*   i_src = (const int*)d_in[7];
    const int*   i_dst = (const int*)d_in[8];
    const float* i_w   = (const float*)d_in[9];
    const float* W_fu  = (const float*)d_in[10];
    const float* b_fu  = (const float*)d_in[11];
    const float* W_fi  = (const float*)d_in[12];
    const float* b_fi  = (const float*)d_in[13];
    const float* W_uc  = (const float*)d_in[14];
    const float* W_ic  = (const float*)d_in[15];
    const float* bn_fu = (const float*)d_in[16];
    const float* bn_hu = (const float*)d_in[17];
    const float* bn_fi = (const float*)d_in[18];
    const float* bn_hi = (const float*)d_in[19];
    const float* W2_fu = (const float*)d_in[20];
    const float* W2_hu = (const float*)d_in[21];
    const float* W2_fi = (const float*)d_in[22];
    const float* W2_hi = (const float*)d_in[23];
    const float* Wdec  = (const float*)d_in[24];
    const float* Wcomb = (const float*)d_in[25];

    const int Bsz = in_sizes[2];                 // 8192
    const int E   = in_sizes[4] / CLASSES;       // 500000
    float* out = (float*)d_out;

    // ---- workspace carve-up (floats) ----
    float* ws = (float*)d_ws;
    size_t o = 0;
    float* agg_u  = ws + o;  o += (size_t)CLASSES * Bsz * D_FEAT;   // reused as decoder tmp
    float* agg_i  = ws + o;  o += (size_t)CLASSES * Bsz * D_FEAT;
    float* f_user = ws + o;  o += (size_t)Bsz * D_RAW;
    float* f_item = ws + o;  o += (size_t)Bsz * D_RAW;
    float* h_user = ws + o;  o += (size_t)Bsz * H_DIM;
    float* h_item = ws + o;  o += (size_t)Bsz * H_DIM;
    float* u_emb  = ws + o;  o += (size_t)Bsz * D_EMB;
    float* i_emb  = ws + o;  o += (size_t)Bsz * D_EMB;
    float* kern   = ws + o;  o += (size_t)CLASSES * D_EMB * D_EMB;
    float* sc_fu  = ws + o;  o += D_RAW;
    float* sh_fu  = ws + o;  o += D_RAW;
    float* sc_fi  = ws + o;  o += D_RAW;
    float* sh_fi  = ws + o;  o += D_RAW;
    float* sc_hu  = ws + o;  o += H_DIM;
    float* sh_hu  = ws + o;  o += H_DIM;
    float* sc_hi  = ws + o;  o += H_DIM;
    float* sh_hi  = ws + o;  o += H_DIM;

    // ---- 1) fold BN (+bias) into per-column affine ----
    bn_prep<<<(D_RAW + 255) / 256, 256, 0, stream>>>(bn_fu, b_fu, D_RAW, sc_fu, sh_fu);
    bn_prep<<<(D_RAW + 255) / 256, 256, 0, stream>>>(bn_fi, b_fi, D_RAW, sc_fi, sh_fi);
    bn_prep<<<(H_DIM + 255) / 256, 256, 0, stream>>>(bn_hu, nullptr, H_DIM, sc_hu, sh_hu);
    bn_prep<<<(H_DIM + 255) / 256, 256, 0, stream>>>(bn_hi, nullptr, H_DIM, sc_hi, sh_hi);

    // ---- 2) zero aggregation buffers (graph-replay safe: every call) ----
    {
        const long long n = 2LL * CLASSES * Bsz * D_FEAT;
        zero_fill<<<2048, 256, 0, stream>>>(agg_u, n);   // agg_u/agg_i contiguous
    }

    // ---- 3) weighted edge scatter-aggregation (2 edges/wave + prefetch) ----
    {
        const long long tw = (long long)CLASSES * E;
        const long long pairs = (tw + 1) / 2;
        const int blocks = (int)((pairs + 7) / 8);
        edge_agg<<<blocks, 256, 0, stream>>>(item_feat, u_src, u_dst, u_w, agg_u, E, Bsz, tw);
        edge_agg<<<blocks, 256, 0, stream>>>(user_feat, i_src, i_dst, i_w, agg_i, E, Bsz, tw);
    }

    // ---- 4) f = relu(bn(gather(feat) @ W + b))  (gather + BN fused) ----
    {
        dim3 g(D_RAW / BN, Bsz / BM);
        gemm_bf16_wmma<<<g, 256, 0, stream>>>(user_feat, D_FEAT, user_idx, W_fu, D_RAW,
                                              f_user, D_RAW, D_FEAT, sc_fu, sh_fu, GF_RELU);
        gemm_bf16_wmma<<<g, 256, 0, stream>>>(item_feat, D_FEAT, item_idx, W_fi, D_RAW,
                                              f_item, D_RAW, D_FEAT, sc_fi, sh_fi, GF_RELU);
    }

    // ---- 5) h[:, c*128:(c+1)*128] = relu(bn(agg[c] @ W_c))  (strided concat) ----
    for (int c = 0; c < CLASSES; ++c) {
        dim3 g(D_CONV / BN, Bsz / BM);   // (1, 64)
        gemm_bf16_wmma<<<g, 256, 0, stream>>>(agg_u + (size_t)c * Bsz * D_FEAT, D_FEAT, nullptr,
                                              W_uc + (size_t)c * D_FEAT * D_CONV, D_CONV,
                                              h_user + (size_t)c * D_CONV, H_DIM, D_FEAT,
                                              sc_hu + c * D_CONV, sh_hu + c * D_CONV, GF_RELU);
        gemm_bf16_wmma<<<g, 256, 0, stream>>>(agg_i + (size_t)c * Bsz * D_FEAT, D_FEAT, nullptr,
                                              W_ic + (size_t)c * D_FEAT * D_CONV, D_CONV,
                                              h_item + (size_t)c * D_CONV, H_DIM, D_FEAT,
                                              sc_hi + c * D_CONV, sh_hi + c * D_CONV, GF_RELU);
    }

    // ---- 6) emb = relu(f @ W2_f + h @ W2_h)  (split-K accumulate) ----
    {
        dim3 g(D_EMB / BN, Bsz / BM);
        gemm_bf16_wmma<<<g, 256, 0, stream>>>(f_user, D_RAW, nullptr, W2_fu, D_EMB,
                                              u_emb, D_EMB, D_RAW, nullptr, nullptr, 0);
        gemm_bf16_wmma<<<g, 256, 0, stream>>>(h_user, H_DIM, nullptr, W2_hu, D_EMB,
                                              u_emb, D_EMB, H_DIM, nullptr, nullptr, GF_ACC | GF_RELU);
        gemm_bf16_wmma<<<g, 256, 0, stream>>>(f_item, D_RAW, nullptr, W2_fi, D_EMB,
                                              i_emb, D_EMB, D_RAW, nullptr, nullptr, 0);
        gemm_bf16_wmma<<<g, 256, 0, stream>>>(h_item, H_DIM, nullptr, W2_hi, D_EMB,
                                              i_emb, D_EMB, H_DIM, nullptr, nullptr, GF_ACC | GF_RELU);
    }

    // ---- 7) bilinear decoder ----
    dec_combine<<<(CLASSES * D_EMB * D_EMB + 255) / 256, 256, 0, stream>>>(Wdec, Wcomb, kern);

    for (int r = 0; r < CLASSES; ++r) {          // tmp_r = u_emb @ kern[r]  (reuse agg_u)
        dim3 g(D_EMB / BN, Bsz / BM);
        gemm_bf16_wmma<<<g, 256, 0, stream>>>(u_emb, D_EMB, nullptr,
                                              kern + (size_t)r * D_EMB * D_EMB, D_EMB,
                                              agg_u + (size_t)r * Bsz * D_EMB, D_EMB,
                                              D_EMB, nullptr, nullptr, 0);
    }
    {
        const long long tw = (long long)CLASSES * Bsz;
        bilinear_dot<<<(int)((tw + 7) / 8), 256, 0, stream>>>(agg_u, i_emb, out, Bsz);
    }
}
